// SEQencoder_3238405341406
// MI455X (gfx1250) — matrix operand order repaired and checked by reading, not compiled, not run on hardware
//
#include <hip/hip_runtime.h>

#define B_  32
#define T_  2048
#define V_  64
#define H_  256
#define HD_ 128
#define G3_ 384   // 3*HD

typedef __attribute__((ext_vector_type(16))) __bf16 v16bf;
typedef __attribute__((ext_vector_type(8)))  __bf16 v8bf;
typedef __attribute__((ext_vector_type(8)))  float  v8f;

// v_rcp_f32-based activations: avoids v_div_scale/div_fmas sequences.
__device__ __forceinline__ float fast_sigmoid(float x) {
    return __builtin_amdgcn_rcpf(1.f + __expf(-x));
}
__device__ __forceinline__ float fast_tanh(float x) {
    float e = __expf(2.f * x);
    return 1.f - 2.f * __builtin_amdgcn_rcpf(e + 1.f);
}

// One block per direction (0 = forward, 1 = backward). 256 threads = 8 waves.
// Wave w owns output columns 16w..16w+15 of ALL THREE gates (N-tiles w, w+8,
// w+16): the gate update happens in registers on the freshly computed
// accumulators. h is register-resident (f32) + double-buffered bf16 in LDS
// for the next step's A-fragments. 1 barrier per step.
__global__ __launch_bounds__(256, 1)
void bigru_wmma_kernel(const int*   __restrict__ tokens,
                       const int*   __restrict__ lengths,
                       const float* __restrict__ emb,
                       const float* __restrict__ Wih_f, const float* __restrict__ Whh_f,
                       const float* __restrict__ bih_f, const float* __restrict__ bhh_f,
                       const float* __restrict__ Wih_b, const float* __restrict__ Whh_b,
                       const float* __restrict__ bih_b, const float* __restrict__ bhh_b,
                       float* __restrict__ out)
{
    const int dir = blockIdx.x;
    const float* Wih = dir ? Wih_b : Wih_f;
    const float* Whh = dir ? Whh_b : Whh_f;
    const float* bih = dir ? bih_b : bih_f;
    const float* bhh = dir ? bhh_b : bhh_f;

    __shared__ float  GI[V_ * G3_];          // 98304 B : emb @ Wih^T + bih
    __shared__ __bf16 HBF[2][B_][HD_ + 8];   // 17408 B : h in bf16, double-buffered
    __shared__ float  BHH[G3_];
    __shared__ int    TOK[2][B_];            // double-buffered per-step tokens
    __shared__ int    LEN[B_];

    const int tid    = threadIdx.x;
    const int lane   = tid & 31;
    const int w      = tid >> 5;     // wave id 0..7
    const int lanelo = lane & 15;
    const int hi     = lane >> 4;
    const int hi8    = hi << 3;
    const int colR   = w * 16 + lanelo;   // this wave's h-column (0..127)

    // ---------------- setup ----------------
    for (int c = tid; c < G3_; c += 256) BHH[c] = bhh[c];
    if (tid < B_) { int l = lengths[tid]; LEN[tid] = (l < 1) ? 1 : l; }
    for (int idx = tid; idx < 2 * B_ * (HD_ + 8); idx += 256)
        ((__bf16*)HBF)[idx] = (__bf16)0.f;

    // GI lookup table: [V, 3HD] = emb @ Wih^T + bih (done once; tiny)
    for (int idx = tid; idx < V_ * G3_; idx += 256) {
        int v = idx / G3_, c = idx - v * G3_;
        const float4* e4 = (const float4*)(emb + v * H_);
        const float4* w4 = (const float4*)(Wih + c * H_);
        float s = bih[c];
        #pragma unroll 4
        for (int k = 0; k < H_ / 4; ++k) {
            float4 a = e4[k], b = w4[k];
            s += a.x * b.x + a.y * b.y + a.z * b.z + a.w * b.w;
        }
        GI[idx] = s;
    }

    // Preload B fragments (Whh^T) into VGPRs, resident for all 2048 steps.
    // Gate g (0=r,1=z,2=n): output column col = colR + g*128.
    // B 32x16 bf16 layout: lane -> N = lane%16; element e -> K = (e<8 ? e : e+8) + 8*(lane>=16).
    v16bf bfrag[3][4];
    #pragma unroll
    for (int g = 0; g < 3; ++g) {
        int col = colR + g * HD_;
        #pragma unroll
        for (int kt = 0; kt < 4; ++kt) {
            #pragma unroll
            for (int e = 0; e < 16; ++e) {
                int k = kt * 32 + (e < 8 ? e : e + 8) + 8 * hi;
                bfrag[g][kt][e] = (__bf16)Whh[col * HD_ + k];
            }
        }
    }
    __syncthreads();   // LEN visible before stager caches it

    // Wave 0 is the token stager: one token per lane, wave-uniform branch.
    const bool is_stager = (__builtin_amdgcn_readfirstlane(w) == 0);
    const int  st_len    = LEN[lane];              // stager's length (lane-indexed)
    const int* st_tokp   = tokens + lane * T_;     // stager's token row
    if (is_stager) {
        int idx = dir ? (st_len - 1) : 0;          // tokens for t = 0
        TOK[0][lane] = st_tokp[idx];
    }

    // Persistent per-element state, C/D register layout:
    // element (mt, v) of this lane -> row b = mt*16 + v + 8*hi, column colR.
    float h_reg[2][8], pool[2][8];
    int   len_reg[2][8];
    #pragma unroll
    for (int mt = 0; mt < 2; ++mt)
        #pragma unroll
        for (int v = 0; v < 8; ++v) {
            h_reg[mt][v] = 0.f;
            pool[mt][v]  = 0.f;
            len_reg[mt][v] = LEN[mt * 16 + v + hi8];
        }
    const float bR = BHH[colR];
    const float bZ = BHH[colR + HD_];
    const float bN = BHH[colR + 2 * HD_];
    __syncthreads();

    // ---------------- recurrence: 1 barrier per step ----------------
    for (int t = 0; t < T_; ++t) {
        const int pb = t & 1;

        // token-row offsets (GI rows) for this step's batch rows
        int tkrow[2][8];
        #pragma unroll
        for (int mt = 0; mt < 2; ++mt)
            #pragma unroll
            for (int v = 0; v < 8; ++v)
                tkrow[mt][v] = TOK[pb][mt * 16 + v + hi8] * G3_;

        // Accumulator init: branch-free, role known statically per gate.
        v8f acc[3][2];   // [gate][mt]
        #pragma unroll
        for (int mt = 0; mt < 2; ++mt)
            #pragma unroll
            for (int v = 0; v < 8; ++v) {
                acc[0][mt][v] = GI[tkrow[mt][v] + colR] + bR;         // gi_r + bhh_r
                acc[1][mt][v] = GI[tkrow[mt][v] + colR + HD_] + bZ;   // gi_z + bhh_z
                acc[2][mt][v] = bN;                                   // bhh_n only
            }

        // K loop: load A fragments per kt (h bf16 from LDS), then 6 independent
        // WMMAs per kt (3 gates x 2 M-tiles) -> dependent WMMAs are 5 apart.
        // 16-bit A 16x32 layout: half-wave hi holds K={8hi..8hi+7, 16+8hi..23+8hi}.
        #pragma unroll
        for (int kt = 0; kt < 4; ++kt) {
            v16bf afrag[2];
            #pragma unroll
            for (int mt = 0; mt < 2; ++mt) {
                int row = mt * 16 + lanelo;
                int k0  = kt * 32 + 8 * hi;
                v8bf lo  = *(const v8bf*)(&HBF[pb][row][k0]);
                v8bf hiv = *(const v8bf*)(&HBF[pb][row][k0 + 16]);
                afrag[mt] = __builtin_shufflevector(lo, hiv,
                    0, 1, 2, 3, 4, 5, 6, 7, 8, 9, 10, 11, 12, 13, 14, 15);
            }
            #pragma unroll
            for (int g = 0; g < 3; ++g)
                #pragma unroll
                for (int mt = 0; mt < 2; ++mt)
                    acc[g][mt] = __builtin_amdgcn_wmma_f32_16x16x32_bf16(
                                     false, afrag[mt], false, bfrag[g][kt],
                                     (short)0, acc[g][mt], false, false);
        }

        // Stage tokens for t+1 (other buffer): wave-uniform, 1 vmem + 1 ds op.
        if (is_stager && t + 1 < T_) {
            int idx = dir ? (st_len - 2 - t) : (t + 1);
            idx = idx < 0 ? 0 : idx;
            TOK[pb ^ 1][lane] = st_tokp[idx];
        }

        // Gate update entirely in registers; write bf16 h to the other buffer.
        #pragma unroll
        for (int mt = 0; mt < 2; ++mt)
            #pragma unroll
            for (int v = 0; v < 8; ++v) {
                float gin  = GI[tkrow[mt][v] + colR + 2 * HD_];       // gi_n
                float r    = fast_sigmoid(acc[0][mt][v]);
                float z    = fast_sigmoid(acc[1][mt][v]);
                float n    = fast_tanh(gin + r * acc[2][mt][v]);
                float hold = h_reg[mt][v];
                float hnew = (1.f - z) * n + z * hold;
                bool  m    = (t < len_reg[mt][v]);
                float h    = m ? hnew : hold;
                h_reg[mt][v] = h;
                pool[mt][v] += m ? h : 0.f;
                HBF[pb ^ 1][mt * 16 + v + hi8][colR] = (__bf16)h;
            }

        __syncthreads();
    }

    // ---------------- write pooled output [B, 2*HD], order (fwd, bwd) ----------------
    #pragma unroll
    for (int mt = 0; mt < 2; ++mt)
        #pragma unroll
        for (int v = 0; v < 8; ++v) {
            int row = mt * 16 + v + hi8;
            float invlen = __builtin_amdgcn_rcpf((float)len_reg[mt][v]);
            out[row * H_ + dir * HD_ + colR] = pool[mt][v] * invlen;
        }
}

extern "C" void kernel_launch(void* const* d_in, const int* in_sizes, int n_in,
                              void* d_out, int out_size, void* d_ws, size_t ws_size,
                              hipStream_t stream) {
    (void)in_sizes; (void)n_in; (void)out_size; (void)d_ws; (void)ws_size;
    bigru_wmma_kernel<<<2, 256, 0, stream>>>(
        (const int*)d_in[0],   // tokens
        (const int*)d_in[1],   // lengths
        (const float*)d_in[2], // emb
        (const float*)d_in[3], (const float*)d_in[4],   // Wih_f, Whh_f
        (const float*)d_in[5], (const float*)d_in[6],   // bih_f, bhh_f
        (const float*)d_in[7], (const float*)d_in[8],   // Wih_b, Whh_b
        (const float*)d_in[9], (const float*)d_in[10],  // bih_b, bhh_b
        (float*)d_out);
}
